// self_att_3951369912398
// MI455X (gfx1250) — matrix operand order repaired and checked by reading, not compiled, not run on hardware
//
#include <hip/hip_runtime.h>
#include <stdint.h>

#define B_  8
#define T_  1024
#define D_  512
#define H_  8
#define HD_ 64
#define EPS_ 1e-5f

typedef __attribute__((ext_vector_type(16))) __bf16 v16bf;
typedef __attribute__((ext_vector_type(8)))  float  v8f;

union Frag16 {
    v16bf v;
    unsigned short us[16];
    uint4 q[2];
};

__device__ __forceinline__ unsigned short f2bf(float f) {
    unsigned u = __builtin_bit_cast(unsigned, f);
    u += 0x7fffu + ((u >> 16) & 1u);           // round-to-nearest-even
    return (unsigned short)(u >> 16);
}
__device__ __forceinline__ float bf2f(unsigned short h) {
    unsigned u = ((unsigned)h) << 16;
    return __builtin_bit_cast(float, u);
}

#if __has_builtin(__builtin_amdgcn_tanhf)
__device__ __forceinline__ float fast_tanh(float x) { return __builtin_amdgcn_tanhf(x); }
#elif __has_builtin(__builtin_amdgcn_tanh_f32)
__device__ __forceinline__ float fast_tanh(float x) { return __builtin_amdgcn_tanh_f32(x); }
#else
__device__ __forceinline__ float fast_tanh(float x) { return tanhf(x); }
#endif

__device__ __forceinline__ v8f wmma_bf16(const Frag16& a, const Frag16& b, v8f c) {
    return __builtin_amdgcn_wmma_f32_16x16x32_bf16(
        false, a.v, false, b.v, (short)0, c, false, false);
}

// Load a B fragment (32x16 K-major) from a TRANSPOSED matrix BT[N,K] row-major:
// lane ll owns column n = ncol; needs BT[n*ldk + k0 + hi*8 + 0..7] and BT[n*ldk + k0 + 16 + hi*8 + 0..7].
__device__ __forceinline__ Frag16 load_bfragT(const unsigned short* __restrict__ BT,
                                              size_t ncol, int ldk, int k0, int hi) {
    Frag16 f;
    const uint4* p = (const uint4*)(BT + ncol * (size_t)ldk + k0 + hi * 8);
    f.q[0] = p[0];
    f.q[1] = p[2];
    return f;
}

// ---------------- fp32 -> bf16 convert ----------------
__global__ void cvt_bf16_kernel(const float* __restrict__ src,
                                unsigned short* __restrict__ dst, int n) {
    int i = blockIdx.x * blockDim.x + threadIdx.x;
    if (i < n) dst[i] = f2bf(src[i]);
}

// fp32 [K,N] -> bf16 transposed [N,K]
__global__ void cvt_bf16_t_kernel(const float* __restrict__ src,
                                  unsigned short* __restrict__ dst, int K, int N) {
    int i = blockIdx.x * blockDim.x + threadIdx.x;
    if (i < K * N) {
        int k = i / N, n = i - k * N;
        dst[(size_t)n * K + k] = f2bf(src[i]);
    }
}

// ---------------- bf16 GEMM: C = bf16(A@B + bias), B given transposed ----------
// A: [M,K] bf16 row-major; BT: [N,K] bf16 row-major.
// Each wave computes one 16x64 output tile; 8 waves / block.
// If CT != nullptr, also store C transposed per batch: CT[(b*N + n)*T_ + t]
// (b = row/1024, t = row%1024) -- used to feed attention B fragments.
__global__ void __launch_bounds__(256)
gemm_bf16_kernel(const unsigned short* __restrict__ A, int lda,
                 const unsigned short* __restrict__ BT,
                 const float* __restrict__ bias,
                 unsigned short* __restrict__ C, int ldc,
                 unsigned short* __restrict__ CT,
                 int M, int N, int K)
{
    const int lane = threadIdx.x & 31;
    const int wave = threadIdx.x >> 5;
    const int ll = lane & 15, hi = lane >> 4;
    const int tilesPerRow = N / 64;
    const int tile = blockIdx.x * 8 + wave;
    const int m0 = (tile / tilesPerRow) * 16;
    const int n0 = (tile % tilesPerRow) * 64;
    if (m0 >= M) return;

    const unsigned short* arow = A + (size_t)(m0 + ll) * lda;

    v8f acc0 = {}, acc1 = {}, acc2 = {}, acc3 = {};
    for (int k0 = 0; k0 < K; k0 += 32) {
        Frag16 a;
        const uint4* ap = (const uint4*)(arow + k0 + hi * 8);
        a.q[0] = ap[0];
        a.q[1] = ap[2];                       // +16 shorts = +2 uint4
        __builtin_prefetch(arow + k0 + 32, 0, 1);
        Frag16 b0 = load_bfragT(BT, n0 + ll,      K, k0, hi);
        Frag16 b1 = load_bfragT(BT, n0 + ll + 16, K, k0, hi);
        Frag16 b2 = load_bfragT(BT, n0 + ll + 32, K, k0, hi);
        Frag16 b3 = load_bfragT(BT, n0 + ll + 48, K, k0, hi);
        acc0 = wmma_bf16(a, b0, acc0);
        acc1 = wmma_bf16(a, b1, acc1);
        acc2 = wmma_bf16(a, b2, acc2);
        acc3 = wmma_bf16(a, b3, acc3);
    }

    float bv0 = bias[n0 + ll], bv1 = bias[n0 + ll + 16];
    float bv2 = bias[n0 + ll + 32], bv3 = bias[n0 + ll + 48];
    unsigned short t0[8], t1[8], t2[8], t3[8];
#pragma unroll
    for (int i = 0; i < 8; ++i) {
        t0[i] = f2bf(acc0[i] + bv0);
        t1[i] = f2bf(acc1[i] + bv1);
        t2[i] = f2bf(acc2[i] + bv2);
        t3[i] = f2bf(acc3[i] + bv3);
    }
#pragma unroll
    for (int i = 0; i < 8; ++i) {
        int row = m0 + i + 8 * hi;
        unsigned short* crow = C + (size_t)row * ldc + n0 + ll;
        crow[0] = t0[i]; crow[16] = t1[i]; crow[32] = t2[i]; crow[48] = t3[i];
    }
    if (CT) {   // packed 8-wide transposed stores (consecutive t per lane)
        int b = m0 >> 10;
        int tb = (m0 & 1023) + 8 * hi;
        *(uint4*)(CT + ((size_t)b * N + n0 + ll     ) * T_ + tb) = *(const uint4*)t0;
        *(uint4*)(CT + ((size_t)b * N + n0 + ll + 16) * T_ + tb) = *(const uint4*)t1;
        *(uint4*)(CT + ((size_t)b * N + n0 + ll + 32) * T_ + tb) = *(const uint4*)t2;
        *(uint4*)(CT + ((size_t)b * N + n0 + ll + 48) * T_ + tb) = *(const uint4*)t3;
    }
}

// ---------------- per-(b,t,h) MLP score scalars ----------------
__global__ void scores_kernel(const unsigned short* __restrict__ qbf,
                              const unsigned short* __restrict__ kbf,
                              const float* __restrict__ w_mlp,
                              float* __restrict__ q_s, float* __restrict__ k_s)
{
    int idx = blockIdx.x * blockDim.x + threadIdx.x;   // bt*H + h
    if (idx >= B_ * T_ * H_) return;
    int h = idx & (H_ - 1);
    int bt = idx >> 3;
    const unsigned short* qrow = qbf + (size_t)bt * D_ + h * HD_;
    const unsigned short* krow = kbf + (size_t)bt * D_ + h * HD_;
    float sq = 0.f, sk = 0.f;
#pragma unroll 8
    for (int d = 0; d < HD_; ++d) {
        sk += bf2f(krow[d]) * w_mlp[d];
        sq += bf2f(qrow[d]) * w_mlp[HD_ + d];
    }
    q_s[idx] = sq;
    k_s[idx] = sk;
}

// ---------------- attention: out = softmax_k(tanh(qs+ks)) @ kx ----------------
// grid.x = (B*H)*4 ; 8 waves/block ; each wave owns TWO 16-row q-blocks.
// kT: [B, 512(=h*64+d), T] bf16  -> B fragments are two b128 loads each.
__global__ void __launch_bounds__(256)
attn_kernel(const unsigned short* __restrict__ kT,
            const float* __restrict__ q_s,
            const float* __restrict__ k_s,
            unsigned short* __restrict__ att)     // [B*T, 512] bf16
{
    __shared__ float ks_sh[T_];

    const int lane = threadIdx.x & 31;
    const int wave = threadIdx.x >> 5;
    const int ll = lane & 15, hi = lane >> 4;
    const int bh  = blockIdx.x >> 2;
    const int seg = blockIdx.x & 3;
    const int b = bh >> 3, h = bh & 7;
    const int qb0 = (seg * 8 + wave) * 2;

    // stage k_s row for this (b,h) into LDS
    for (int i = threadIdx.x; i < T_; i += 256)
        ks_sh[i] = k_s[((size_t)b * T_ + i) * H_ + h];
    __syncthreads();

    const float qsA = q_s[((size_t)b * T_ + qb0 * 16 + ll) * H_ + h];
    const float qsB = q_s[((size_t)b * T_ + qb0 * 16 + 16 + ll) * H_ + h];

    Frag16 ones;
#pragma unroll
    for (int j = 0; j < 16; ++j) ones.us[j] = 0x3F80;   // bf16 1.0

    const unsigned short* kTb = kT + ((size_t)b * D_ + h * HD_) * T_;

    v8f accA0 = {}, accA1 = {}, accA2 = {}, accA3 = {}, zA = {};
    v8f accB0 = {}, accB1 = {}, accB2 = {}, accB3 = {}, zB = {};

    for (int k0 = 0; k0 < T_; k0 += 32) {
        Frag16 aA, aB;
#pragma unroll
        for (int j = 0; j < 8; ++j) {
            float ks0 = ks_sh[k0 + hi * 8 + j];
            float ks1 = ks_sh[k0 + 16 + hi * 8 + j];
            // tanh in [-1,1] -> exp in [1/e, e]: single-pass softmax is safe
            aA.us[j]     = f2bf(__expf(fast_tanh(qsA + ks0)));
            aA.us[8 + j] = f2bf(__expf(fast_tanh(qsA + ks1)));
            aB.us[j]     = f2bf(__expf(fast_tanh(qsB + ks0)));
            aB.us[8 + j] = f2bf(__expf(fast_tanh(qsB + ks1)));
        }
        Frag16 b0 = load_bfragT(kTb, ll,      T_, k0, hi);
        Frag16 b1 = load_bfragT(kTb, ll + 16, T_, k0, hi);
        Frag16 b2 = load_bfragT(kTb, ll + 32, T_, k0, hi);
        Frag16 b3 = load_bfragT(kTb, ll + 48, T_, k0, hi);
        __builtin_prefetch(kTb + (size_t)ll * T_ + k0 + 32, 0, 1);

        accA0 = wmma_bf16(aA, b0, accA0);
        accA1 = wmma_bf16(aA, b1, accA1);
        accA2 = wmma_bf16(aA, b2, accA2);
        accA3 = wmma_bf16(aA, b3, accA3);
        zA    = wmma_bf16(aA, ones, zA);     // row-sums -> softmax denominator
        accB0 = wmma_bf16(aB, b0, accB0);
        accB1 = wmma_bf16(aB, b1, accB1);
        accB2 = wmma_bf16(aB, b2, accB2);
        accB3 = wmma_bf16(aB, b3, accB3);
        zB    = wmma_bf16(aB, ones, zB);
    }

#pragma unroll
    for (int i = 0; i < 8; ++i) {
        int rowA = b * T_ + qb0 * 16 + i + 8 * hi;
        int rowB = rowA + 16;
        float rzA = 1.0f / zA[i];
        float rzB = 1.0f / zB[i];
        unsigned short* ca = att + (size_t)rowA * D_ + h * HD_ + ll;
        unsigned short* cb = att + (size_t)rowB * D_ + h * HD_ + ll;
        ca[0]  = f2bf(accA0[i] * rzA);
        ca[16] = f2bf(accA1[i] * rzA);
        ca[32] = f2bf(accA2[i] * rzA);
        ca[48] = f2bf(accA3[i] * rzA);
        cb[0]  = f2bf(accB0[i] * rzB);
        cb[16] = f2bf(accB1[i] * rzB);
        cb[32] = f2bf(accB2[i] * rzB);
        cb[48] = f2bf(accB3[i] * rzB);
    }
}

// ---------------- att @ Wp + bp, residual, LayerNorm ----------------
// grid.x = M/16; 8 waves each compute 16x64 of the same 16-row block.
__global__ void __launch_bounds__(256)
proj_ln_kernel(const unsigned short* __restrict__ att,   // [8192,512] bf16
               const unsigned short* __restrict__ WpT,   // [512,512] bf16 (transposed)
               const float* __restrict__ bp,
               const float* __restrict__ x,
               const float* __restrict__ ln_g,
               const float* __restrict__ ln_b,
               float* __restrict__ out)
{
    __shared__ float ytile[16][D_ + 4];
    __shared__ float red[2][16][16];
    __shared__ float mu_s[16], rs_s[16];

    const int lane = threadIdx.x & 31;
    const int wave = threadIdx.x >> 5;
    const int ll = lane & 15, hi = lane >> 4;
    const int m0 = blockIdx.x * 16;
    const int n0 = wave * 64;

    const unsigned short* arow = att + (size_t)(m0 + ll) * D_;

    v8f acc0 = {}, acc1 = {}, acc2 = {}, acc3 = {};
    for (int k0 = 0; k0 < D_; k0 += 32) {
        Frag16 a;
        const uint4* ap = (const uint4*)(arow + k0 + hi * 8);
        a.q[0] = ap[0];
        a.q[1] = ap[2];
        Frag16 b0 = load_bfragT(WpT, n0 + ll,      D_, k0, hi);
        Frag16 b1 = load_bfragT(WpT, n0 + ll + 16, D_, k0, hi);
        Frag16 b2 = load_bfragT(WpT, n0 + ll + 32, D_, k0, hi);
        Frag16 b3 = load_bfragT(WpT, n0 + ll + 48, D_, k0, hi);
        acc0 = wmma_bf16(a, b0, acc0);
        acc1 = wmma_bf16(a, b1, acc1);
        acc2 = wmma_bf16(a, b2, acc2);
        acc3 = wmma_bf16(a, b3, acc3);
    }
#pragma unroll
    for (int i = 0; i < 8; ++i) {
        int r = i + 8 * hi;
        int row = m0 + r;
        int c = n0 + ll;
        ytile[r][c]      = x[(size_t)row * D_ + c]      + acc0[i] + bp[c];
        ytile[r][c + 16] = x[(size_t)row * D_ + c + 16] + acc1[i] + bp[c + 16];
        ytile[r][c + 32] = x[(size_t)row * D_ + c + 32] + acc2[i] + bp[c + 32];
        ytile[r][c + 48] = x[(size_t)row * D_ + c + 48] + acc3[i] + bp[c + 48];
    }
    __syncthreads();
    {
        int r = threadIdx.x >> 4;
        int t = threadIdx.x & 15;
        float s = 0.f, ss = 0.f;
        for (int j = t; j < D_; j += 16) { float v = ytile[r][j]; s += v; ss += v * v; }
        red[0][r][t] = s; red[1][r][t] = ss;
    }
    __syncthreads();
    if (threadIdx.x < 16) {
        int r = threadIdx.x;
        float s = 0.f, ss = 0.f;
#pragma unroll
        for (int j = 0; j < 16; ++j) { s += red[0][r][j]; ss += red[1][r][j]; }
        float mu = s * (1.0f / D_);
        float var = ss * (1.0f / D_) - mu * mu;
        mu_s[r] = mu;
        rs_s[r] = rsqrtf(var + EPS_);
    }
    __syncthreads();
    {
        int r = threadIdx.x >> 4;
        int t = threadIdx.x & 15;
        float mu = mu_s[r], rs = rs_s[r];
        for (int j = t; j < D_; j += 16) {
            float v = (ytile[r][j] - mu) * rs;
            out[(size_t)(m0 + r) * D_ + j] = v * ln_g[j] + ln_b[j];
        }
    }
}

extern "C" void kernel_launch(void* const* d_in, const int* in_sizes, int n_in,
                              void* d_out, int out_size, void* d_ws, size_t ws_size,
                              hipStream_t stream) {
    const float* x     = (const float*)d_in[0];
    const float* Wq    = (const float*)d_in[1];
    const float* bq    = (const float*)d_in[2];
    const float* Wk    = (const float*)d_in[3];
    const float* bk    = (const float*)d_in[4];
    const float* Wp    = (const float*)d_in[5];
    const float* bp    = (const float*)d_in[6];
    const float* w_mlp = (const float*)d_in[7];
    const float* ln_g  = (const float*)d_in[8];
    const float* ln_b  = (const float*)d_in[9];
    float* out = (float*)d_out;

    const int BT = B_ * T_;               // 8192
    char* ws = (char*)d_ws;
    size_t off = 0;
    unsigned short* x_bf   = (unsigned short*)(ws + off); off += (size_t)BT * D_ * 2;
    unsigned short* wqT_bf = (unsigned short*)(ws + off); off += (size_t)D_ * D_ * 2;
    unsigned short* wkT_bf = (unsigned short*)(ws + off); off += (size_t)D_ * D_ * 2;
    unsigned short* wpT_bf = (unsigned short*)(ws + off); off += (size_t)D_ * D_ * 2;
    unsigned short* q_bf   = (unsigned short*)(ws + off); off += (size_t)BT * D_ * 2;
    unsigned short* k_bf   = (unsigned short*)(ws + off); off += (size_t)BT * D_ * 2;
    unsigned short* kT_bf  = (unsigned short*)(ws + off); off += (size_t)BT * D_ * 2;
    unsigned short* a_bf   = (unsigned short*)(ws + off); off += (size_t)BT * D_ * 2;
    float* qs = (float*)(ws + off); off += (size_t)BT * H_ * 4;
    float* ks = (float*)(ws + off); off += (size_t)BT * H_ * 4;
    (void)ws_size; (void)in_sizes; (void)n_in; (void)out_size;

    // 1) converts (weights transposed for contiguous B-fragment loads)
    {
        int n = BT * D_;
        cvt_bf16_kernel<<<(n + 255) / 256, 256, 0, stream>>>(x, x_bf, n);
        int nw = D_ * D_;
        cvt_bf16_t_kernel<<<(nw + 255) / 256, 256, 0, stream>>>(Wq, wqT_bf, D_, D_);
        cvt_bf16_t_kernel<<<(nw + 255) / 256, 256, 0, stream>>>(Wk, wkT_bf, D_, D_);
        cvt_bf16_t_kernel<<<(nw + 255) / 256, 256, 0, stream>>>(Wp, wpT_bf, D_, D_);
    }
    // 2) q/k projections; k also emits transposed copy for attention B operand
    gemm_bf16_kernel<<<512, 256, 0, stream>>>(x_bf, D_, wqT_bf, bq, q_bf, D_,
                                              nullptr, BT, D_, D_);
    gemm_bf16_kernel<<<512, 256, 0, stream>>>(x_bf, D_, wkT_bf, bk, k_bf, D_,
                                              kT_bf, BT, D_, D_);
    // 3) q_s / k_s scalars
    {
        int n = BT * H_;
        scores_kernel<<<(n + 255) / 256, 256, 0, stream>>>(q_bf, k_bf, w_mlp, qs, ks);
    }
    // 4) attention
    attn_kernel<<<B_ * H_ * 4, 256, 0, stream>>>(kT_bf, qs, ks, a_bf);
    // 5) projection + residual + LayerNorm
    proj_ln_kernel<<<BT / 16, 256, 0, stream>>>(a_bf, wpT_bf, bp, x, ln_g, ln_b, out);
}
